// SelectiveSSMKernel_60902636257785
// MI455X (gfx1250) — matrix operand order repaired and checked by reading, not compiled, not run on hardware
//
#include <hip/hip_runtime.h>
#include <hip/hip_bf16.h>
#include <stddef.h>

// ---------------------------------------------------------------------------
// Selective SSM (diagonal A) via chunked SSD reformulation on gfx1250.
// B=8, D=1024, N=64, L=2048, chunk Q=64.
//
// Kernel 1 (per batch x chunk): Bu/Cu = [B;C]^T u  via bf16 WMMA, then fused
//   computation of the causal intra-chunk kernel M and the decay-weighted
//   bf16 operands consumed by kernel 2.  Done once per (b,c), in parallel.
// Kernel 2 (per batch x d-tile): serial 32-chunk scan carrying h in WMMA
//   accumulators; workspace operands double-buffered into LDS with
//   global_load_async_to_lds_b128 + s_wait_asynccnt (issued one chunk ahead).
// ---------------------------------------------------------------------------

#define BATCH   8
#define DMODEL  1024
#define NSTATE  64
#define SEQL    2048
#define Q       64
#define NCHUNK  (SEQL / Q)   // 32
#define CHELEM  (NSTATE * Q) // 4096 elements per chunk matrix

typedef __attribute__((ext_vector_type(16))) __bf16 v16bf;
typedef __attribute__((ext_vector_type(8)))  float  v8f;

__device__ __forceinline__ __bf16 f2bf(float x) {
    unsigned u = __builtin_bit_cast(unsigned, x);
    unsigned r = (u + 0x7FFFu + ((u >> 16) & 1u)) >> 16;   // round-nearest-even
    unsigned short s = (unsigned short)r;
    return __builtin_bit_cast(__bf16, s);
}

// CDNA5 16-bit A-matrix (16x32) K index for half h (0..15) in lane `lane`.
__device__ __forceinline__ int kidx(int lane, int h) {
    return (h & 7) + ((h & 8) ? 16 : 0) + ((lane & 16) ? 8 : 0);
}

// gfx1250 async copy: 16 bytes per lane, global -> LDS, tracked by ASYNCcnt.
__device__ __forceinline__ void async_g2l_b128(unsigned lds_byte_off, const void* gaddr) {
    asm volatile("global_load_async_to_lds_b128 %0, %1, off"
                 :: "v"(lds_byte_off), "v"(gaddr) : "memory");
}
__device__ __forceinline__ void wait_async_le6() {
    asm volatile("s_wait_asynccnt 6" ::: "memory");
}
__device__ __forceinline__ void wait_async_0() {
    asm volatile("s_wait_asynccnt 0" ::: "memory");
}

// ---------------------------------------------------------------------------
// Kernel 1: projections + fused M.  One workgroup per (chunk, batch).
// Outputs (all bf16, laid out row-contiguous for kernel2's async row copies):
//   BuWq[b][c][n][j] = Bu[n,j] * a_n^(63-j)
//   CuAq[b][c][n][l] = Cu[n,l] * a_n^(l+1)
//   Mtq [b][c][j][l] = (j<=l) ? sum_n Cu[n,l] Bu[n,j] a_n^(l-j) : 0
// ---------------------------------------------------------------------------
__global__ __launch_bounds__(256) void ssm_proj_kernel(
    const float* __restrict__ U,     // (B, D, L)
    const float* __restrict__ Bm,    // (D, N)
    const float* __restrict__ Cm,    // (D, N)
    const float* __restrict__ Adiag, // (N, N)
    __bf16* __restrict__ BuWq,
    __bf16* __restrict__ CuAq,
    __bf16* __restrict__ Mtq)
{
    const int c    = blockIdx.x;
    const int b    = blockIdx.y;
    const int tid  = threadIdx.x;
    const int lane = tid & 31;
    const int wave = tid >> 5;
    const int l0   = c * Q;

    __shared__ __bf16 Wl[128][34];   // [m][k]  m: 0..63 B-rows(n), 64..127 C-rows
    __shared__ __bf16 Ul[32][72];    // [k][l]
    __shared__ float  P[64][68];     // P[n][t] = a_n^t, t = 0..64
    __shared__ float  Sp[128][66];   // f32 spill of [Bu;Cu]

    if (tid < 64) {
        float a = Adiag[(size_t)tid * NSTATE + tid];
        float x = 1.0f;
        for (int t = 0; t <= 64; ++t) { P[tid][t] = x; x *= a; }
    }

    v8f acc[4] = {};
    const int mrow = wave * 16 + (lane & 15);
    const int krow = (lane & 15) + ((lane >> 4) << 4);

    for (int d0 = 0; d0 < DMODEL; d0 += 32) {
        __syncthreads();
        for (int idx = tid; idx < 32 * 64; idx += 256) {
            int k = idx >> 6, n = idx & 63;
            Wl[n][k]      = f2bf(Bm[(size_t)(d0 + k) * NSTATE + n]);
            Wl[64 + n][k] = f2bf(Cm[(size_t)(d0 + k) * NSTATE + n]);
        }
        for (int idx = tid; idx < 32 * 64; idx += 256) {
            int k = idx >> 6, j = idx & 63;
            Ul[k][j] = f2bf(U[((size_t)b * DMODEL + d0 + k) * SEQL + l0 + j]);
        }
        __syncthreads();

        v16bf af;
        #pragma unroll
        for (int h = 0; h < 16; ++h) af[h] = Wl[mrow][kidx(lane, h)];
        #pragma unroll
        for (int nt = 0; nt < 4; ++nt) {
            v16bf bfr;
            #pragma unroll
            for (int h = 0; h < 16; ++h) bfr[h] = Ul[kidx(lane, h)][nt * 16 + (lane & 15)];
            acc[nt] = __builtin_amdgcn_wmma_f32_16x16x32_bf16(
                false, af, false, bfr, (short)0, acc[nt], false, false);
        }
    }

    // spill f32 [Bu;Cu] + store decay-weighted bf16 operands for kernel2
    const size_t gb = (size_t)(b * NCHUNK + c) * CHELEM;
    #pragma unroll
    for (int nt = 0; nt < 4; ++nt) {
        #pragma unroll
        for (int r = 0; r < 8; ++r) {
            int m = wave * 16 + r + ((lane >> 4) << 3);
            int j = nt * 16 + (lane & 15);
            float v = acc[nt][r];
            Sp[m][j] = v;
            if (m < 64) BuWq[gb + (size_t)m * Q + j]        = f2bf(v * P[m][63 - j]);
            else        CuAq[gb + (size_t)(m - 64) * Q + j] = f2bf(v * P[m - 64][j + 1]);
        }
    }
    __syncthreads();

    // causal intra-chunk kernel, once per (b,c); coalesced bf16 stores over l
    for (int idx = tid; idx < 64 * 64; idx += 256) {
        int j = idx >> 6, l = idx & 63;
        float s = 0.0f;
        if (j <= l) {
            #pragma unroll 4
            for (int n = 0; n < 64; ++n)
                s += Sp[64 + n][l] * Sp[n][j] * P[n][l - j];
        }
        Mtq[gb + (size_t)j * Q + l] = f2bf(s);
    }
}

// ---------------------------------------------------------------------------
// Kernel 2: chunked scan.  One workgroup per (d-tile of 64, batch).
//   y = U_chunk * M^T + h_in^T * CuA ;  h <- a^Q .* h + BuW * U_chunk^T
// Workspace operands (BuW, CuA, Mt) stream in via async global->LDS copies,
// double buffered one chunk ahead.
// ---------------------------------------------------------------------------
__global__ __launch_bounds__(256) void ssm_scan_kernel(
    const float*  __restrict__ U,      // (B, D, L)
    const float*  __restrict__ Adiag,  // (N, N)
    const __bf16* __restrict__ BuWq,
    const __bf16* __restrict__ CuAq,
    const __bf16* __restrict__ Mtq,
    float* __restrict__ Y)             // (B, D, L)
{
    const int dt0  = blockIdx.x;
    const int b    = blockIdx.y;
    const int d0   = dt0 * 64;
    const int tid  = threadIdx.x;
    const int lane = tid & 31;
    const int wave = tid >> 5;
    const int wr   = wave & 3;          // tile row
    const int wc   = (wave >> 2) * 2;   // tile col pair base

    __shared__ __bf16 Ubf[64][72];          // u chunk [d][j]
    __shared__ __bf16 Hbf[64][72];          // h spill [n][d]
    __shared__ __bf16 Buf[2][3][64][72];    // double buffer {BuW, CuA, Mt}
    __shared__ float  aQs[64];              // a_n^64

    if (tid < 64) {
        float x = Adiag[(size_t)tid * NSTATE + tid];
        #pragma unroll
        for (int t = 0; t < 6; ++t) x *= x;  // a^64
        aQs[tid] = x;
    }

    const __bf16* gsrc[3] = {BuWq, CuAq, Mtq};

    // issue one chunk of async copies into buffer `par`
    auto issue_chunk = [&](int par, int cc) {
        const size_t gb = (size_t)(b * NCHUNK + cc) * CHELEM;
        #pragma unroll
        for (int t = 0; t < 3; ++t) {
            const char* gbase = (const char*)(gsrc[t] + gb);
            #pragma unroll
            for (int s = 0; s < 2; ++s) {
                int seg = tid * 2 + s;        // 512 x 16B segments = 8 KB
                int row = seg >> 3, col = seg & 7;
                unsigned loff = (unsigned)(size_t)&Buf[par][t][row][col * 8];
                async_g2l_b128(loff, gbase + row * 128 + col * 16);
            }
        }
    };

    issue_chunk(0, 0);                  // prefetch chunk 0
    v8f h[2] = {};
    __syncthreads();                    // aQs visible

    const int mrow = wr * 16 + (lane & 15);
    const int krow = (lane & 15) + ((lane >> 4) << 4);

    for (int c = 0; c < NCHUNK; ++c) {
        const int cur = c & 1;
        const int l0  = c * Q;

        if (c + 1 < NCHUNK) issue_chunk(1 - cur, c + 1);   // one chunk ahead

        // stage u chunk (f32 -> bf16); overlaps with in-flight async copies
        for (int idx = tid; idx < 64 * 64; idx += 256) {
            int r = idx >> 6, q = idx & 63;
            Ubf[r][q] = f2bf(U[((size_t)b * DMODEL + d0 + r) * SEQL + l0 + q]);
        }
        // spill h_in for the y_inter GEMM (transpose access through LDS)
        #pragma unroll
        for (int t = 0; t < 2; ++t)
            #pragma unroll
            for (int r = 0; r < 8; ++r)
                Hbf[wr * 16 + r + ((lane >> 4) << 3)][(wc + t) * 16 + (lane & 15)] =
                    f2bf(h[t][r]);

        if (c + 1 < NCHUNK) wait_async_le6();   // FIFO: current chunk done
        else                wait_async_0();
        __syncthreads();

        __bf16 (*BW)[72] = Buf[cur][0];
        __bf16 (*CA)[72] = Buf[cur][1];
        __bf16 (*MT)[72] = Buf[cur][2];

        // ---- y = U*M^T + h_in^T*CuA : two 16x16 tiles per wave ----
        v8f y[2] = {};
        #pragma unroll
        for (int ks = 0; ks < 64; ks += 32) {
            v16bf afu, afh;
            #pragma unroll
            for (int hh = 0; hh < 16; ++hh) {
                afu[hh] = Ubf[mrow][kidx(lane, hh) + ks];      // A: u   (m=d, k=j)
                afh[hh] = Hbf[kidx(lane, hh) + ks][mrow];      // A: h^T (m=d, k=n)
            }
            #pragma unroll
            for (int t = 0; t < 2; ++t) {
                v16bf bm, bc;
                #pragma unroll
                for (int hh = 0; hh < 16; ++hh) {
                    bm[hh] = MT[kidx(lane, hh) + ks][(wc + t) * 16 + (lane & 15)];  // B: M^T (k=j, n=l)
                    bc[hh] = CA[kidx(lane, hh) + ks][(wc + t) * 16 + (lane & 15)];  // B: CuA (k=n, n=l)
                }
                y[t] = __builtin_amdgcn_wmma_f32_16x16x32_bf16(
                    false, afu, false, bm, (short)0, y[t], false, false);
                y[t] = __builtin_amdgcn_wmma_f32_16x16x32_bf16(
                    false, afh, false, bc, (short)0, y[t], false, false);
            }
        }
        #pragma unroll
        for (int t = 0; t < 2; ++t)
            #pragma unroll
            for (int r = 0; r < 8; ++r) {
                int d = d0 + wr * 16 + r + ((lane >> 4) << 3);
                int l = l0 + (wc + t) * 16 + (lane & 15);
                Y[((size_t)b * DMODEL + d) * SEQL + l] = y[t][r];
            }

        // ---- h <- a^Q .* h + BuW * U^T ----
        #pragma unroll
        for (int t = 0; t < 2; ++t)
            #pragma unroll
            for (int r = 0; r < 8; ++r)
                h[t][r] *= aQs[wr * 16 + r + ((lane >> 4) << 3)];
        #pragma unroll
        for (int ks = 0; ks < 64; ks += 32) {
            v16bf af;
            #pragma unroll
            for (int hh = 0; hh < 16; ++hh)
                af[hh] = BW[mrow][kidx(lane, hh) + ks];        // A: BuW (m=n, k=j)
            #pragma unroll
            for (int t = 0; t < 2; ++t) {
                v16bf bu;
                #pragma unroll
                for (int hh = 0; hh < 16; ++hh)
                    bu[hh] = Ubf[(wc + t) * 16 + (lane & 15)][kidx(lane, hh) + ks]; // B: u^T (k=j, n=d)
                h[t] = __builtin_amdgcn_wmma_f32_16x16x32_bf16(
                    false, af, false, bu, (short)0, h[t], false, false);
            }
        }
        __syncthreads();   // protect Ubf/Hbf/Buf[cur] before next iteration
    }
}

// ---------------------------------------------------------------------------
// Host launch.  Inputs: d_in[0]=L(int), d_in[1]=u, d_in[2]=A, d_in[3]=B,
// d_in[4]=C (f32).  d_out = y (B, D, L) f32.  Workspace: 6 MB of bf16 chunks.
// ---------------------------------------------------------------------------
extern "C" void kernel_launch(void* const* d_in, const int* in_sizes, int n_in,
                              void* d_out, int out_size, void* d_ws, size_t ws_size,
                              hipStream_t stream) {
    (void)in_sizes; (void)n_in; (void)out_size; (void)ws_size;
    const float* U  = (const float*)d_in[1];
    const float* A  = (const float*)d_in[2];
    const float* Bm = (const float*)d_in[3];
    const float* Cm = (const float*)d_in[4];
    float* Y = (float*)d_out;

    __bf16* BuWq = (__bf16*)d_ws;                                  // 2 MB
    __bf16* CuAq = BuWq + (size_t)BATCH * NCHUNK * CHELEM;         // 2 MB
    __bf16* Mtq  = CuAq + (size_t)BATCH * NCHUNK * CHELEM;         // 2 MB

    ssm_proj_kernel<<<dim3(NCHUNK, BATCH), 256, 0, stream>>>(U, Bm, Cm, A, BuWq, CuAq, Mtq);
    ssm_scan_kernel<<<dim3(DMODEL / 64, BATCH), 256, 0, stream>>>(U, A, BuWq, CuAq, Mtq, Y);
}
